// MambaCell_8538394984487
// MI455X (gfx1250) — matrix operand order repaired and checked by reading, not compiled, not run on hardware
//
#include <hip/hip_runtime.h>
#include <hip/hip_bf16.h>
#include <cstdint>
#include <cstddef>

// ---- types ----------------------------------------------------------------
typedef __attribute__((ext_vector_type(16))) _Float16 v16h;
typedef __attribute__((ext_vector_type(8)))  _Float16 v8h;
typedef __attribute__((ext_vector_type(4)))  _Float16 v4h;
typedef __attribute__((ext_vector_type(8)))  float    v8f;
typedef __attribute__((ext_vector_type(4)))  float    v4f;
typedef __attribute__((ext_vector_type(4)))  unsigned v4u;

#define D_MODEL 1024
#define D_STATE 16
#define BATCH   4096
#define N_ALL   (2*D_MODEL + D_MODEL*D_STATE*2)   // 34816 = u | delta | bc

union FragAB { v16h h; v4u q[2]; };

// Two b128 async DMAs global->LDS, 16B apart (same imm offset applies to both
// the LDS and the global side per ISA 10.7 pseudocode).
#define ASYNC_LOAD2(ldsoff, gaddr)                                        \
  asm volatile("global_load_async_to_lds_b128 %0, %1, off\n\t"            \
               "global_load_async_to_lds_b128 %0, %1, off offset:16"      \
               :: "v"(ldsoff), "v"(gaddr) : "memory")

#define WAIT_ASYNC() asm volatile("s_wait_asynccnt 0x0" ::: "memory")

// ---- fp32 -> fp16 conversion (vectorized, grid-stride) --------------------
__global__ __launch_bounds__(256)
void cvt_f32_to_f16(const float* __restrict__ src, _Float16* __restrict__ dst, int n4) {
  int i = blockIdx.x * blockDim.x + threadIdx.x;
  int stride = gridDim.x * blockDim.x;
  for (; i < n4; i += stride) {
    v4f v = ((const v4f*)src)[i];
    ((v4h*)dst)[i] = __builtin_convertvector(v, v4h);
  }
}

// ---- WMMA f16 GEMM:  C[M,N] = A[M,K] * B[N,K]^T  (K = 1024) ---------------
// Block tile 128x128, 8 waves (4 m-waves x 2 n-waves), each wave = 32x64
// (2x4 accumulators -> 8 WMMAs per 32-deep K step).
// Global->LDS staging via double-buffered GLOBAL_LOAD_ASYNC_TO_LDS_B128.
__global__ __launch_bounds__(256)
void wmma_gemm_nt(const _Float16* __restrict__ A,    // [M x 1024] f16
                  const _Float16* __restrict__ Bm,   // [N x 1024] f16
                  _Float16* __restrict__ outh,       // f16 output (or null)
                  float*    __restrict__ outf,       // f32 output (or null)
                  const float* __restrict__ bias,    // optional, for f32 path
                  int N)                             // output row stride / cols
{
  constexpr int K   = 1024;
  constexpr int BK  = 32;
  constexpr int SA  = 40;                 // padded LDS row stride (halves)
  constexpr unsigned BUFB = 128 * SA * 2; // bytes per buffer (10240)

  __shared__ _Float16 lsA[2][128 * SA];
  __shared__ _Float16 lsB[2][128 * SA];

  const int t    = threadIdx.x;
  const int lane = t & 31;
  const int w    = t >> 5;
  const int bm0  = blockIdx.y * 128;
  const int bn0  = blockIdx.x * 128;
  const int mw   = (w & 3) * 32;          // wave row offset in block tile
  const int nw   = (w >> 2) * 64;         // wave col offset in block tile

  // async staging map: thread t moves 16 contiguous halves of row (t>>1),
  // starting at half-column (t&1)*16, for both A and B tiles (128x32 each)
  const int srow = t >> 1;
  const int scol = (t & 1) * 16;
  const unsigned ldsA0 = (unsigned)(size_t)&lsA[0][srow * SA + scol];
  const unsigned ldsB0 = (unsigned)(size_t)&lsB[0][srow * SA + scol];
  const uint64_t gA = (uint64_t)(size_t)(A  + (size_t)(bm0 + srow) * K + scol);
  const uint64_t gB = (uint64_t)(size_t)(Bm + (size_t)(bn0 + srow) * K + scol);

  const int g = lane >> 4;                // half-wave group (0/1)
  const int l15 = lane & 15;

  v8f acc[2][4];
#pragma unroll
  for (int i = 0; i < 2; ++i)
#pragma unroll
    for (int j = 0; j < 4; ++j) acc[i][j] = (v8f){};

  // prologue: DMA tile 0 into buffer 0
  ASYNC_LOAD2(ldsA0, gA);
  ASYNC_LOAD2(ldsB0, gB);
  WAIT_ASYNC();
  __syncthreads();

  int buf = 0;
  for (int kk = 0; kk < K; kk += BK) {
    const int nxt = kk + BK;
    if (nxt < K) {                         // prefetch next tile into other buf
      const unsigned la = ldsA0 + (unsigned)(buf ^ 1) * BUFB;
      const unsigned lb = ldsB0 + (unsigned)(buf ^ 1) * BUFB;
      ASYNC_LOAD2(la, gA + (uint64_t)(nxt * 2));
      ASYNC_LOAD2(lb, gB + (uint64_t)(nxt * 2));
    }

    const _Float16* bA = &lsA[buf][0];
    const _Float16* bB = &lsB[buf][0];

    // A fragments (16x32 f16): lane m = lane&15; v0..3 -> K = 8g+0..7,
    // v4..7 -> K = 16+8g+0..7 (two contiguous 16B reads)
    FragAB fa[2];
#pragma unroll
    for (int i = 0; i < 2; ++i) {
      const int mrow = mw + i * 16 + l15;
      fa[i].q[0] = *(const v4u*)&bA[mrow * SA + 8 * g];
      fa[i].q[1] = *(const v4u*)&bA[mrow * SA + 8 * g + 16];
    }
    // B fragments (32x16 f16): lane n = lane&15; v0..3 -> K = 8g+0..7,
    // v4..7 -> K = 16+8g+0..7
    FragAB fb[4];
#pragma unroll
    for (int j = 0; j < 4; ++j) {
      const int nrow = nw + j * 16 + l15;
      fb[j].q[0] = *(const v4u*)&bB[nrow * SA + 8 * g];
      fb[j].q[1] = *(const v4u*)&bB[nrow * SA + 8 * g + 16];
    }

#pragma unroll
    for (int i = 0; i < 2; ++i)
#pragma unroll
      for (int j = 0; j < 4; ++j)
        acc[i][j] = __builtin_amdgcn_wmma_f32_16x16x32_f16(
            false, fa[i].h, false, fb[j].h, (short)0, acc[i][j], false, false);

    WAIT_ASYNC();                          // prefetch (overlapped) done
    __syncthreads();                       // everyone finished reading buf
    buf ^= 1;
  }

  // C/D layout: VGPR r, lanes 0-15 -> M=r, lanes 16-31 -> M=8+r; N = lane&15
#pragma unroll
  for (int i = 0; i < 2; ++i) {
#pragma unroll
    for (int j = 0; j < 4; ++j) {
#pragma unroll
      for (int r = 0; r < 8; ++r) {
        const int grow = bm0 + mw + i * 16 + 8 * g + r;
        const int gcol = bn0 + nw + j * 16 + l15;
        if (outh) {
          outh[(size_t)grow * N + gcol] = (_Float16)acc[i][j][r];
        } else {
          const float b0 = bias ? bias[gcol] : 0.f;
          outf[(size_t)grow * N + gcol] = acc[i][j][r] + b0;
        }
      }
    }
  }
}

// ---- pointwise SSM update --------------------------------------------------
__global__ __launch_bounds__(256)
void ssm_pointwise(const _Float16* __restrict__ P,      // [B x N_ALL] f16
                   const float* __restrict__ h,         // [B x 1024 x 16]
                   const float* __restrict__ A_log,     // [1024 x 16]
                   const float* __restrict__ Dp,        // [1024]
                   const float* __restrict__ dt_b,      // [1024]
                   float* __restrict__ h_new,           // [B x 1024 x 16]
                   _Float16* __restrict__ yact)         // [B x 1024] f16
{
  const int idx = blockIdx.x * 256 + threadIdx.x;       // 0 .. B*D-1
  const int b   = idx >> 10;
  const int ch  = idx & (D_MODEL - 1);

  const _Float16* row = P + (size_t)b * N_ALL;
  const float u  = (float)row[ch];
  const float dr = (float)row[D_MODEL + ch] + dt_b[ch];
  const float delta = (dr > 20.f) ? dr : log1pf(__expf(dr));

  const _Float16* bc = row + 2 * D_MODEL + ch * 32;     // 16 B + 16 C halves
  v8h Bv0 = *(const v8h*)(bc);
  v8h Bv1 = *(const v8h*)(bc + 8);
  v8h Cv0 = *(const v8h*)(bc + 16);
  v8h Cv1 = *(const v8h*)(bc + 24);
  float Bt[16], Ct[16];
#pragma unroll
  for (int j = 0; j < 8; ++j) {
    Bt[j] = (float)Bv0[j]; Bt[j + 8] = (float)Bv1[j];
    Ct[j] = (float)Cv0[j]; Ct[j + 8] = (float)Cv1[j];
  }

  const v4f* hp = (const v4f*)(h     + ((size_t)b * D_MODEL + ch) * D_STATE);
  v4f*       ho = (v4f*)      (h_new + ((size_t)b * D_MODEL + ch) * D_STATE);
  const v4f* ap = (const v4f*)(A_log + (size_t)ch * D_STATE);

  float y = 0.f;
#pragma unroll
  for (int q = 0; q < 4; ++q) {
    v4f hv = hp[q];
    v4f av = ap[q];
    v4f hn;
#pragma unroll
    for (int e = 0; e < 4; ++e) {
      const int j = q * 4 + e;
      const float Aj = -__expf(av[e]);
      const float dA = __expf(delta * Aj);
      const float v  = dA * hv[e] + delta * Bt[j] * u;
      hn[e] = v;
      y += v * Ct[j];
    }
    ho[q] = hn;
  }
  y += Dp[ch] * u;
  const float s = y * (1.f / (1.f + __expf(-y)));       // silu
  yact[idx] = (_Float16)s;
}

// ---- launcher --------------------------------------------------------------
extern "C" void kernel_launch(void* const* d_in, const int* in_sizes, int n_in,
                              void* d_out, int out_size, void* d_ws, size_t ws_size,
                              hipStream_t stream) {
  const float* x     = (const float*)d_in[0];
  const float* h     = (const float*)d_in[1];
  const float* in_w  = (const float*)d_in[2];
  const float* dt_w  = (const float*)d_in[3];
  const float* dt_b  = (const float*)d_in[4];
  const float* bc_w  = (const float*)d_in[5];
  const float* A_log = (const float*)d_in[6];
  const float* Dp    = (const float*)d_in[7];
  const float* out_w = (const float*)d_in[8];
  const float* out_b = (const float*)d_in[9];

  float* out   = (float*)d_out;                          // [4096 x 1024]
  float* h_new = out + (size_t)BATCH * D_MODEL;          // [4096 x 1024 x 16]

  // workspace layout (bytes)
  char* ws = (char*)d_ws;
  _Float16* xh    = (_Float16*)(ws + 0);                 //   8 MB: x in f16
  _Float16* w_all = (_Float16*)(ws + 8388608);           //  68 MB: [34816 x 1024]
  _Float16* w_out = (_Float16*)(ws + 79691776);          //   2 MB: [1024 x 1024]
  _Float16* P     = (_Float16*)(ws + 81788928);          // 272 MB: [4096 x 34816]
  _Float16* yact  = (_Float16*)(ws + 367001600);         //   8 MB: [4096 x 1024]

  // 1) convert x + weights to f16 (weights packed: in_proj | dt_proj | bc_proj)
  auto launch_cvt = [&](const float* s, _Float16* d, int nelem) {
    int n4 = nelem / 4;
    int blocks = (n4 + 255) / 256;
    if (blocks > 16384) blocks = 16384;
    cvt_f32_to_f16<<<blocks, 256, 0, stream>>>(s, d, n4);
  };
  launch_cvt(x,     xh,                          BATCH * D_MODEL);
  launch_cvt(in_w,  w_all,                       D_MODEL * D_MODEL);
  launch_cvt(dt_w,  w_all + 1048576,             D_MODEL * D_MODEL);
  launch_cvt(bc_w,  w_all + 2097152,             2 * D_MODEL * D_STATE * D_MODEL);
  launch_cvt(out_w, w_out,                       D_MODEL * D_MODEL);

  // 2) fused projection GEMM: P[4096, 34816] = xh @ w_all^T   (f16 out)
  dim3 grid1(N_ALL / 128, BATCH / 128);
  wmma_gemm_nt<<<grid1, 256, 0, stream>>>(xh, w_all, P, nullptr, nullptr, N_ALL);

  // 3) pointwise SSM: h_new + silu(y)
  ssm_pointwise<<<(BATCH * D_MODEL) / 256, 256, 0, stream>>>(
      P, h, A_log, Dp, dt_b, h_new, yact);

  // 4) output GEMM: out = yact @ w_out^T + out_b   (f32 out)
  dim3 grid2(D_MODEL / 128, BATCH / 128);
  wmma_gemm_nt<<<grid2, 256, 0, stream>>>(yact, w_out, nullptr, out, out_b, D_MODEL);
}
